// MHCLiteBlock_4020089389292
// MI455X (gfx1250) — matrix operand order repaired and checked by reading, not compiled, not run on hardware
//
#include <hip/hip_runtime.h>
#include <hip/hip_bf16.h>
#include <math.h>

// ---------------- problem constants (fixed by reference) ----------------
#define NSTR 4
#define CDIM 2048
#define NCDIM 8192          // NSTR*CDIM
#define NFACT 24
#define TOUT 32             // 2*N + NFACT
#define DFF 8192
#define NTOK 8192           // B*S
#define EPSV 1.1920929e-07f

#define BK 64
#define LDSP 72             // padded LDS row pitch (bf16 elements): 64 data + 8 pad

typedef __bf16 bf16;
typedef __bf16 v8bf  __attribute__((ext_vector_type(8)));
typedef __bf16 v16bf __attribute__((ext_vector_type(16)));
typedef float  v8f   __attribute__((ext_vector_type(8)));
typedef int    v4i   __attribute__((ext_vector_type(4)));

static __device__ inline v8f zero8() {
    v8f z;
#pragma unroll
    for (int i = 0; i < 8; ++i) z[i] = 0.0f;
    return z;
}

#define WMMA_BF16(a, b, c) \
    __builtin_amdgcn_wmma_f32_16x16x32_bf16(false, (a), false, (b), (short)0, (c), false, false)

// ---------------- async global->LDS staging (gfx1250 path + fallback) ----------------
#if defined(__has_builtin)
#if __has_builtin(__builtin_amdgcn_global_load_async_to_lds_b128) && \
    __has_builtin(__builtin_amdgcn_s_wait_asynccnt)
#define USE_ASYNC_LDS 1
#endif
#endif
#ifndef USE_ASYNC_LDS
#define USE_ASYNC_LDS 0
#endif

#if USE_ASYNC_LDS
typedef __attribute__((address_space(1))) v4i gv4i;  // global int4
typedef __attribute__((address_space(3))) v4i sv4i;  // LDS int4
#define WAIT_ASYNC(n) __builtin_amdgcn_s_wait_asynccnt(n)
static __device__ inline void copy16_g2l(const void* g, void* l) {
    __builtin_amdgcn_global_load_async_to_lds_b128(
        (gv4i*)const_cast<void*>(g), (sv4i*)l, 0, 0);
}
#else
#define WAIT_ASYNC(n)
static __device__ inline void copy16_g2l(const void* g, void* l) {
    *(uint4*)l = *(const uint4*)g;
}
#endif

// Stage one 128 x BK bf16 tile (row-major global, ld elements) into LDS with
// pitch LDSP. 256 threads x 4 x 16B chunks = 128*64*2 bytes.
static __device__ inline void stage_tile(const bf16* __restrict__ gbase, int ld,
                                         int row0, int k0, bf16* sbase, int tid) {
#pragma unroll
    for (int i = 0; i < 4; ++i) {
        int idx = tid + 256 * i;          // 0..1023
        int r = idx >> 3;                 // 0..127
        int j = idx & 7;                  // 16B chunk within row
        const bf16* g = gbase + (size_t)(row0 + r) * ld + k0 + j * 8;
        bf16* l = sbase + r * LDSP + j * 8;
        copy16_g2l(g, l);
    }
}

// ---------------- WMMA fragment loaders ----------------
// A-fragment (16x32 bf16). Lane l: m=l&15, kh=l>>4. elems[0..7]=K[kh*8..+7],
// elems[8..15]=K[16+kh*8..+7] (ISA 7.12.2).
static __device__ inline v16bf frag_a(const bf16* __restrict__ base, int pitch,
                                      int row0, int k0, int lane) {
    int m = lane & 15, kh = lane >> 4;
    const bf16* r = base + (size_t)(row0 + m) * pitch + k0;
    v8bf lo = *(const v8bf*)(r + kh * 8);
    v8bf hi = *(const v8bf*)(r + 16 + kh * 8);
    return __builtin_shufflevector(lo, hi, 0, 1, 2, 3, 4, 5, 6, 7,
                                   8, 9, 10, 11, 12, 13, 14, 15);
}

// B-fragment (32x16 bf16). B[k][n] = W[n0+n][k], W row-major: lane reads row
// (n0 + l&15); elems[0..7]=K[kh*8..+7], elems[8..15]=K[16+kh*8..+7].
static __device__ inline v16bf frag_b(const bf16* __restrict__ base, int pitch,
                                      int nrow0, int k0, int lane) {
    int n = lane & 15, kh = lane >> 4;
    const bf16* r = base + (size_t)(nrow0 + n) * pitch + k0;
    v8bf lo = *(const v8bf*)(r + kh * 8);
    v8bf hi = *(const v8bf*)(r + 16 + kh * 8);
    return __builtin_shufflevector(lo, hi, 0, 1, 2, 3, 4, 5, 6, 7,
                                   8, 9, 10, 11, 12, 13, 14, 15);
}

// A-fragment from fp32 source with per-row scale (RMS norm fused).
static __device__ inline v16bf frag_a_f32_scaled(const float* __restrict__ base, int ld,
                                                 int row0, int k0, float scale, int lane) {
    int m = lane & 15, kh = lane >> 4;
    const float* r = base + (size_t)(row0 + m) * ld + k0;
    v16bf a;
#pragma unroll
    for (int i = 0; i < 8; ++i) a[i] = (bf16)(r[kh * 8 + i] * scale);
#pragma unroll
    for (int i = 0; i < 8; ++i) a[8 + i] = (bf16)(r[16 + kh * 8 + i] * scale);
    return a;
}

// ---------------- K0: fp32 -> bf16 weight conversion ----------------
__global__ void cvt_bf16_kernel(const float* __restrict__ s, bf16* __restrict__ d, int n) {
    int i = blockIdx.x * 256 + threadIdx.x;
    if (i < n) d[i] = (bf16)s[i];
}

// ---------------- K1: per-token RMS scale ----------------
__global__ __launch_bounds__(256) void rms_kernel(const float* __restrict__ x,
                                                  float* __restrict__ rnorm) {
    __shared__ float red[256];
    int t = blockIdx.x;
    const float* row = x + (size_t)t * NCDIM;
    float ss = 0.0f;
    for (int k = threadIdx.x * 4; k < NCDIM; k += 256 * 4) {
        float4 v = *(const float4*)(row + k);
        ss += v.x * v.x + v.y * v.y + v.z * v.z + v.w * v.w;
    }
    red[threadIdx.x] = ss;
    __syncthreads();
    for (int off = 128; off > 0; off >>= 1) {
        if (threadIdx.x < off) red[threadIdx.x] += red[threadIdx.x + off];
        __syncthreads();
    }
    if (threadIdx.x == 0)
        rnorm[t] = rsqrtf(red[0] * (1.0f / (float)NCDIM) + EPSV);
}

// ---------------- K2: projection (WMMA) + gating epilogue ----------------
__global__ __launch_bounds__(32) void proj_gate_kernel(
    const float* __restrict__ x, const bf16* __restrict__ Wab,
    const float* __restrict__ b_all, const float* __restrict__ a_pre,
    const float* __restrict__ a_post, const float* __restrict__ a_res,
    const float* __restrict__ perm, const float* __restrict__ rnorm,
    float* __restrict__ hpre, float* __restrict__ hpost, float* __restrict__ Hm) {
    int t0 = blockIdx.x * 16;
    int lane = threadIdx.x;
    float r = rnorm[t0 + (lane & 15)];

    v8f acc0 = zero8(), acc1 = zero8();
    for (int kb = 0; kb < NCDIM; kb += 32) {
        v16bf a  = frag_a_f32_scaled(x, NCDIM, t0, kb, r, lane);
        v16bf b0 = frag_b(Wab, NCDIM, 0, kb, lane);
        v16bf b1 = frag_b(Wab, NCDIM, 16, kb, lane);
        acc0 = WMMA_BF16(a, b0, acc0);
        acc1 = WMMA_BF16(a, b1, acc1);
    }

    __shared__ float sp[16][33];
    int n = lane & 15, hh = lane >> 4;
#pragma unroll
    for (int v = 0; v < 8; ++v) {
        sp[v + 8 * hh][n]      = acc0[v];
        sp[v + 8 * hh][16 + n] = acc1[v];
    }
    __syncthreads();

    if (lane < 16) {
        int t = t0 + lane;
        float p[TOUT];
#pragma unroll
        for (int o = 0; o < TOUT; ++o) p[o] = sp[lane][o];
        float ap = a_pre[0], aq = a_post[0], ar = a_res[0];
        float hp[NSTR], hq[NSTR];
#pragma unroll
        for (int i = 0; i < NSTR; ++i)
            hp[i] = 1.0f / (1.0f + __expf(-(ap * p[i] + b_all[i])));
#pragma unroll
        for (int i = 0; i < NSTR; ++i)
            hq[i] = 2.0f / (1.0f + __expf(-(aq * p[NSTR + i] + b_all[NSTR + i])));
        float lg[NFACT], mx = -1e30f;
#pragma unroll
        for (int i = 0; i < NFACT; ++i) {
            lg[i] = ar * p[2 * NSTR + i] + b_all[2 * NSTR + i];
            mx = fmaxf(mx, lg[i]);
        }
        float s = 0.0f;
#pragma unroll
        for (int i = 0; i < NFACT; ++i) { lg[i] = __expf(lg[i] - mx); s += lg[i]; }
        float inv = 1.0f / s;
#pragma unroll
        for (int i = 0; i < NSTR; ++i) {
            hpre[t * NSTR + i]  = hp[i];
            hpost[t * NSTR + i] = hq[i];
        }
#pragma unroll
        for (int e = 0; e < 16; ++e) {
            float acc = 0.0f;
#pragma unroll
            for (int i = 0; i < NFACT; ++i) acc += lg[i] * perm[i * 16 + e];
            Hm[t * 16 + e] = acc * inv - hq[e >> 2] * hp[e & 3];
        }
    }
}

// ---------------- K3: layer_input = sum_n h_pre[n] * x[n,:]  (bf16) ----------------
__global__ void layer_input_kernel(const float* __restrict__ x,
                                   const float* __restrict__ hpre,
                                   bf16* __restrict__ li) {
    size_t idx = (size_t)blockIdx.x * 256 + threadIdx.x;  // t*C + c
    int t = (int)(idx / CDIM);
    int c = (int)(idx % CDIM);
    const float* hp = hpre + t * NSTR;
    const float* xr = x + (size_t)t * NCDIM + c;
    float v = hp[0] * xr[0] + hp[1] * xr[CDIM] + hp[2] * xr[2 * CDIM] + hp[3] * xr[3 * CDIM];
    li[idx] = (bf16)v;
}

// ---------------- K4: GEMM1 (tokens x DFF, K=C) + exact GELU -> h (bf16) ----------------
// 128x128 tile, BK=64 double-buffered async LDS staging, 8 waves in 4x2 grid.
__global__ __launch_bounds__(256) void gemm1_gelu_kernel(
    const bf16* __restrict__ A, const bf16* __restrict__ Bw,
    const float* __restrict__ b1, bf16* __restrict__ H) {
    __shared__ bf16 sA[2][128 * LDSP];
    __shared__ bf16 sB[2][128 * LDSP];
    int tid = threadIdx.x;
    int lane = tid & 31, w = tid >> 5;
    int wm = w >> 1, wn = w & 1;
    int m0 = blockIdx.y * 128;
    int n0 = blockIdx.x * 128;

    v8f acc[2][4];
#pragma unroll
    for (int i = 0; i < 2; ++i)
#pragma unroll
        for (int j = 0; j < 4; ++j) acc[i][j] = zero8();

    const int S = CDIM / BK;  // 32 stages
    stage_tile(A, CDIM, m0, 0, sA[0], tid);
    stage_tile(Bw, CDIM, n0, 0, sB[0], tid);
    for (int s = 0; s < S; ++s) {
        if (s + 1 < S) {
            stage_tile(A, CDIM, m0, (s + 1) * BK, sA[(s + 1) & 1], tid);
            stage_tile(Bw, CDIM, n0, (s + 1) * BK, sB[(s + 1) & 1], tid);
            WAIT_ASYNC(8);
        } else {
            WAIT_ASYNC(0);
        }
        __syncthreads();
        const bf16* cA = sA[s & 1];
        const bf16* cB = sB[s & 1];
#pragma unroll
        for (int kt = 0; kt < 2; ++kt) {
            v16bf af[2], bfr[4];
            af[0] = frag_a(cA, LDSP, wm * 32, kt * 32, lane);
            af[1] = frag_a(cA, LDSP, wm * 32 + 16, kt * 32, lane);
#pragma unroll
            for (int j = 0; j < 4; ++j)
                bfr[j] = frag_b(cB, LDSP, wn * 64 + j * 16, kt * 32, lane);
#pragma unroll
            for (int i = 0; i < 2; ++i)
#pragma unroll
                for (int j = 0; j < 4; ++j)
                    acc[i][j] = WMMA_BF16(af[i], bfr[j], acc[i][j]);
        }
        __syncthreads();
    }

    int n = lane & 15, hh = lane >> 4;
#pragma unroll
    for (int i = 0; i < 2; ++i)
#pragma unroll
        for (int j = 0; j < 4; ++j) {
            int f = n0 + wn * 64 + j * 16 + n;
            float bias = b1[f];
#pragma unroll
            for (int v = 0; v < 8; ++v) {
                int t = m0 + wm * 32 + i * 16 + v + 8 * hh;
                float val = acc[i][j][v] + bias;
                float g = 0.5f * val * (1.0f + erff(val * 0.70710678f));
                H[(size_t)t * DFF + f] = (bf16)g;
            }
        }
}

// ---------------- K5: GEMM2 (tokens x C, K=DFF) + fused stream remix ----------------
__global__ __launch_bounds__(256) void gemm2_out_kernel(
    const bf16* __restrict__ Hbuf, const bf16* __restrict__ Bw2,
    const float* __restrict__ b2, const float* __restrict__ x,
    const float* __restrict__ hpost, const float* __restrict__ Hm,
    float* __restrict__ out) {
    __shared__ bf16 sA[2][128 * LDSP];
    __shared__ bf16 sB[2][128 * LDSP];
    int tid = threadIdx.x;
    int lane = tid & 31, w = tid >> 5;
    int wm = w >> 1, wn = w & 1;
    int m0 = blockIdx.y * 128;  // token tile
    int n0 = blockIdx.x * 128;  // channel tile

    v8f acc[2][4];
#pragma unroll
    for (int i = 0; i < 2; ++i)
#pragma unroll
        for (int j = 0; j < 4; ++j) acc[i][j] = zero8();

    const int S = DFF / BK;  // 128 stages
    stage_tile(Hbuf, DFF, m0, 0, sA[0], tid);
    stage_tile(Bw2, DFF, n0, 0, sB[0], tid);
    for (int s = 0; s < S; ++s) {
        if (s + 1 < S) {
            stage_tile(Hbuf, DFF, m0, (s + 1) * BK, sA[(s + 1) & 1], tid);
            stage_tile(Bw2, DFF, n0, (s + 1) * BK, sB[(s + 1) & 1], tid);
            WAIT_ASYNC(8);
        } else {
            WAIT_ASYNC(0);
        }
        __syncthreads();
        const bf16* cA = sA[s & 1];
        const bf16* cB = sB[s & 1];
#pragma unroll
        for (int kt = 0; kt < 2; ++kt) {
            v16bf af[2], bfr[4];
            af[0] = frag_a(cA, LDSP, wm * 32, kt * 32, lane);
            af[1] = frag_a(cA, LDSP, wm * 32 + 16, kt * 32, lane);
#pragma unroll
            for (int j = 0; j < 4; ++j)
                bfr[j] = frag_b(cB, LDSP, wn * 64 + j * 16, kt * 32, lane);
#pragma unroll
            for (int i = 0; i < 2; ++i)
#pragma unroll
                for (int j = 0; j < 4; ++j)
                    acc[i][j] = WMMA_BF16(af[i], bfr[j], acc[i][j]);
        }
        __syncthreads();
    }

    int n = lane & 15, hh = lane >> 4;
#pragma unroll
    for (int i = 0; i < 2; ++i)
#pragma unroll
        for (int j = 0; j < 4; ++j) {
            int c = n0 + wn * 64 + j * 16 + n;
            float bias = b2[c];
#pragma unroll
            for (int v = 0; v < 8; ++v) {
                int t = m0 + wm * 32 + i * 16 + v + 8 * hh;
                float lo = acc[i][j][v] + bias;  // layer_output[t][c]
                const float* xr = x + (size_t)t * NCDIM + c;
                const float* hm = Hm + t * 16;
                const float* hp = hpost + t * NSTR;
                float x0 = xr[0], x1 = xr[CDIM], x2 = xr[2 * CDIM], x3 = xr[3 * CDIM];
#pragma unroll
                for (int ii = 0; ii < NSTR; ++ii) {
                    float val = hm[ii * 4 + 0] * x0 + hm[ii * 4 + 1] * x1 +
                                hm[ii * 4 + 2] * x2 + hm[ii * 4 + 3] * x3 + hp[ii] * lo;
                    out[(size_t)t * NCDIM + (size_t)ii * CDIM + c] = val;
                }
            }
        }
}

// ---------------- host-side launch ----------------
extern "C" void kernel_launch(void* const* d_in, const int* in_sizes, int n_in,
                              void* d_out, int out_size, void* d_ws, size_t ws_size,
                              hipStream_t stream) {
    const float* x     = (const float*)d_in[0];
    const float* W_all = (const float*)d_in[1];
    const float* b_all = (const float*)d_in[2];
    const float* a_pre = (const float*)d_in[3];
    const float* a_post= (const float*)d_in[4];
    const float* a_res = (const float*)d_in[5];
    const float* W1    = (const float*)d_in[6];
    const float* b1    = (const float*)d_in[7];
    const float* W2    = (const float*)d_in[8];
    const float* b2    = (const float*)d_in[9];
    const float* perm  = (const float*)d_in[10];
    float* out = (float*)d_out;

    char* w = (char*)d_ws;
    size_t off = 0;
    auto carve = [&](size_t bytes) {
        void* p = w + off;
        off = (off + bytes + 255) & ~(size_t)255;
        return p;
    };
    bf16* W1bf   = (bf16*)carve((size_t)DFF * CDIM * sizeof(bf16));
    bf16* W2bf   = (bf16*)carve((size_t)CDIM * DFF * sizeof(bf16));
    bf16* Wabf   = (bf16*)carve((size_t)TOUT * NCDIM * sizeof(bf16));
    float* rnorm = (float*)carve((size_t)NTOK * sizeof(float));
    float* hpre  = (float*)carve((size_t)NTOK * NSTR * sizeof(float));
    float* hpost = (float*)carve((size_t)NTOK * NSTR * sizeof(float));
    float* Hm    = (float*)carve((size_t)NTOK * 16 * sizeof(float));
    bf16* li     = (bf16*)carve((size_t)NTOK * CDIM * sizeof(bf16));
    bf16* Hbuf   = (bf16*)carve((size_t)NTOK * DFF * sizeof(bf16));
    (void)ws_size; (void)in_sizes; (void)n_in; (void)out_size;

    {
        int n1 = DFF * CDIM;
        cvt_bf16_kernel<<<(n1 + 255) / 256, 256, 0, stream>>>(W1, W1bf, n1);
        int n2 = CDIM * DFF;
        cvt_bf16_kernel<<<(n2 + 255) / 256, 256, 0, stream>>>(W2, W2bf, n2);
        int n3 = TOUT * NCDIM;
        cvt_bf16_kernel<<<(n3 + 255) / 256, 256, 0, stream>>>(W_all, Wabf, n3);
    }
    rms_kernel<<<NTOK, 256, 0, stream>>>(x, rnorm);
    proj_gate_kernel<<<NTOK / 16, 32, 0, stream>>>(x, Wabf, b_all, a_pre, a_post,
                                                   a_res, perm, rnorm, hpre, hpost, Hm);
    layer_input_kernel<<<(NTOK * CDIM) / 256, 256, 0, stream>>>(x, hpre, li);
    gemm1_gelu_kernel<<<dim3(DFF / 128, NTOK / 128), 256, 0, stream>>>(li, W1bf, b1, Hbuf);
    gemm2_out_kernel<<<dim3(CDIM / 128, NTOK / 128), 256, 0, stream>>>(Hbuf, W2bf, b2, x,
                                                                       hpost, Hm, out);
}
